// GCN_Expert_3109556322394
// MI455X (gfx1250) — matrix operand order repaired and checked by reading, not compiled, not run on hardware
//
#include <hip/hip_runtime.h>
#include <hip/hip_bf16.h>

// ---------------- problem constants (match reference) ----------------
#define NCLS     40
#define FDIM     256
#define FHID     64
#define DHID     32
#define HID      128
#define INCH     136
#define INCH_P   160   // padded to multiple of 32 for WMMA K loop
#define OUTP     48    // 40 padded to 48 (3 column tiles)
#define MAXDEG   256

typedef __bf16 bf16_t;

#if defined(__gfx1250__)
typedef __attribute__((ext_vector_type(16))) __bf16 v16bf;
typedef __attribute__((ext_vector_type(8)))  __bf16 v8bf;
typedef __attribute__((ext_vector_type(8)))  float  v8f;
union ABfrag { v16bf v; v8bf h[2]; };
#endif

__device__ inline bf16_t f2bf(float f) {
    unsigned u = __builtin_bit_cast(unsigned, f);
    unsigned r = u + 0x7FFFu + ((u >> 16) & 1u);   // round-to-nearest-even
    unsigned short h = (unsigned short)(r >> 16);
    return __builtin_bit_cast(bf16_t, h);
}

// ---------------- graph preprocessing ----------------
__global__ void k_degree(const int* __restrict__ src, const int* __restrict__ dst,
                         int* degA, int* degD, int E) {
    int e = blockIdx.x * blockDim.x + threadIdx.x;
    if (e >= E) return;
    atomicAdd(&degA[src[e]], 1);   // deg = deg(src)+deg(dst) appearances
    atomicAdd(&degA[dst[e]], 1);
    atomicAdd(&degD[dst[e]], 1);   // gcn_norm degree (dst column)
}

__global__ void k_dinv(const int* __restrict__ degD, float* __restrict__ dinv, int N) {
    int n = blockIdx.x * blockDim.x + threadIdx.x;
    if (n >= N) return;
    float dg = (float)(degD[n] + 1);   // + self loop, always > 0
    dinv[n] = rsqrtf(dg);
}

// fp32 -> bf16 bulk convert
__global__ void k_f2bf(const float* __restrict__ in, bf16_t* __restrict__ out, long total) {
    long i = (long)blockIdx.x * blockDim.x + threadIdx.x;
    if (i >= total) return;
    out[i] = f2bf(in[i]);
}

// W [Ksrc x Nsrc] fp32 row-major -> WT [Npad x Kpad] bf16 (column-major, zero padded)
__global__ void k_wconv(const float* __restrict__ W, bf16_t* __restrict__ WT,
                        int Ksrc, int Nsrc, int Kpad, int Npad) {
    int i = blockIdx.x * blockDim.x + threadIdx.x;
    int total = Kpad * Npad;
    if (i >= total) return;
    int k = i % Kpad, n = i / Kpad;
    float v = (k < Ksrc && n < Nsrc) ? W[(long)k * Nsrc + n] : 0.0f;
    WT[(long)n * Kpad + k] = f2bf(v);
}

// fill X0 cols 0..39 (logits), 104..135 (deg emb), 136..159 (zero); 40..103 by GEMM
__global__ void k_assemble(const float* __restrict__ logits, const float* __restrict__ deg_table,
                           const int* __restrict__ degA, bf16_t* __restrict__ X0, long total) {
    long i = (long)blockIdx.x * blockDim.x + threadIdx.x;
    if (i >= total) return;
    int n = (int)(i / INCH_P);
    int c = (int)(i % INCH_P);
    if (c < NCLS) {
        X0[i] = f2bf(logits[(long)n * NCLS + c]);
    } else if (c < NCLS + FHID) {
        // written by feature-projection GEMM
    } else if (c < INCH) {
        int dgi = degA[n];
        dgi = dgi < 0 ? 0 : (dgi > MAXDEG - 1 ? MAXDEG - 1 : dgi);
        X0[i] = f2bf(deg_table[(long)dgi * DHID + (c - (NCLS + FHID))]);
    } else {
        X0[i] = f2bf(0.0f);
    }
}

// ---------------- WMMA GEMM, register-blocked over columns ----------------
// One wave per 16-row strip. A fragments (16 x KIN) are loaded ONCE into
// registers, then the wave sweeps all NT column tiles re-streaming only the
// (tiny, L0-resident) column-major weight matrix. rows % 16 == 0 required
// (N = 50000 = 3125*16), EXEC all-1s for WMMA.
template<int KIN, int NT>
__global__ void __launch_bounds__(32)
k_gemm_wide(const bf16_t* __restrict__ A, int sA,
            const bf16_t* __restrict__ BT,      // [NT*16 x KIN] column-major
            const float*  __restrict__ bias,    // nullptr -> no bias
            float* __restrict__ Cf,             // fp32 out (or nullptr)
            bf16_t* __restrict__ Cb,            // bf16 out (used if Cf==nullptr)
            int sC, int coff) {
#if defined(__gfx1250__)
    constexpr int KF = KIN / 32;
    const int lane = threadIdx.x;           // 0..31
    const int m    = lane & 15;
    const int hi   = lane >> 4;             // 0/1
    const long row = (long)blockIdx.x * 16 + m;

    // load all A fragments for this 16-row strip once
    const bf16_t* arow = A + row * sA;
    ABfrag a[KF];
    #pragma unroll
    for (int kf = 0; kf < KF; ++kf) {
        // A 16x32: elems 0..7 -> K = k0+hi*8.. ; elems 8..15 -> K = k0+hi*8+16..
        a[kf].h[0] = *(const v8bf*)(arow + kf * 32 + hi * 8);
        a[kf].h[1] = *(const v8bf*)(arow + kf * 32 + hi * 8 + 16);
    }

    #pragma unroll
    for (int nt = 0; nt < NT; ++nt) {
        const int n = nt * 16 + m;          // B/C column for this lane
        const bf16_t* bcol = BT + (long)n * KIN;
        v8f acc = {};
        #pragma unroll
        for (int kf = 0; kf < KF; ++kf) {
            ABfrag b;
            // B 32x16: elems 0..7 -> K = k0+hi*8.. ; elems 8..15 -> K = k0+hi*8+16.. (column-major W)
            b.h[0] = *(const v8bf*)(bcol + kf * 32 + hi * 8);
            b.h[1] = *(const v8bf*)(bcol + kf * 32 + hi * 8 + 16);
            acc = __builtin_amdgcn_wmma_f32_16x16x32_bf16(false, a[kf].v, false, b.v,
                                                          (short)0, acc, false, false);
        }
        const float bv = bias ? bias[n] : 0.0f;
        #pragma unroll
        for (int v = 0; v < 8; ++v) {
            long r = (long)blockIdx.x * 16 + hi * 8 + v;   // D row for VGPR v
            float val = acc[v] + bv;
            if (Cf) Cf[r * sC + coff + n] = val;
            else    Cb[r * sC + coff + n] = f2bf(val);
        }
    }
#endif
}

// ---------------- edge aggregation ----------------
// F=128: one wave per edge, float4 per lane, 4 fp32 atomics (L2-resident)
__global__ void __launch_bounds__(256)
k_scatter128(const int* __restrict__ src, const int* __restrict__ dst,
             const float* __restrict__ dinv, const float* __restrict__ H,
             float* __restrict__ Agg, int E) {
    long gid = (long)blockIdx.x * blockDim.x + threadIdx.x;
    int e = (int)(gid >> 5);
    int lane = (int)(gid & 31);
    if (e >= E) return;
    int s = src[e], d = dst[e];
    float w = dinv[s] * dinv[d];
    float4 v = *(const float4*)(H + (long)s * HID + lane * 4);
    float* o = Agg + (long)d * HID + lane * 4;
    atomicAdd(o + 0, v.x * w);
    atomicAdd(o + 1, v.y * w);
    atomicAdd(o + 2, v.z * w);
    atomicAdd(o + 3, v.w * w);
}

// last layer: H3 stride 48 (cols 40..47 are zero pad), out stride 40; 10 float4 per edge
__global__ void __launch_bounds__(256)
k_scatter40(const int* __restrict__ src, const int* __restrict__ dst,
            const float* __restrict__ dinv, const float* __restrict__ H3,
            float* __restrict__ out, int E) {
    long gid = (long)blockIdx.x * blockDim.x + threadIdx.x;
    int e = (int)(gid / 10);
    int q = (int)(gid % 10);
    if (e >= E) return;
    int s = src[e], d = dst[e];
    float w = dinv[s] * dinv[d];
    float4 v = *(const float4*)(H3 + (long)s * OUTP + q * 4);
    float* o = out + (long)d * NCLS + q * 4;
    atomicAdd(o + 0, v.x * w);
    atomicAdd(o + 1, v.y * w);
    atomicAdd(o + 2, v.z * w);
    atomicAdd(o + 3, v.w * w);
}

// hidden finalize: self-loop term + bias + relu -> bf16 for next GEMM A
__global__ void k_finalize_hidden(const float* __restrict__ Agg, const float* __restrict__ Hlin,
                                  const float* __restrict__ dinv, const float* __restrict__ b,
                                  bf16_t* __restrict__ Xn, long total) {
    long i = (long)blockIdx.x * blockDim.x + threadIdx.x;
    if (i >= total) return;
    int n = (int)(i >> 7);        // /128
    int f = (int)(i & 127);
    float di = dinv[n];
    float v = Agg[i] + Hlin[i] * di * di + b[f];
    Xn[i] = f2bf(fmaxf(v, 0.0f));
}

// output finalize: d_out already holds edge aggregation; add self-loop + bias (no relu)
__global__ void k_finalize_out(const float* __restrict__ H3, const float* __restrict__ dinv,
                               const float* __restrict__ b, float* __restrict__ out, long total) {
    long i = (long)blockIdx.x * blockDim.x + threadIdx.x;
    if (i >= total) return;
    int n = (int)(i / NCLS);
    int f = (int)(i % NCLS);
    float di = dinv[n];
    out[i] += H3[(long)n * OUTP + f] * di * di + b[f];
}

// ---------------- host launcher ----------------
static inline size_t alignup(size_t x) { return (x + 255) & ~(size_t)255; }

extern "C" void kernel_launch(void* const* d_in, const int* in_sizes, int n_in,
                              void* d_out, int out_size, void* d_ws, size_t ws_size,
                              hipStream_t stream) {
    const float* logits    = (const float*)d_in[0];
    const float* features  = (const float*)d_in[1];
    const int*   edge      = (const int*)  d_in[2];
    const float* Wf        = (const float*)d_in[3];
    const float* bf        = (const float*)d_in[4];
    const float* deg_table = (const float*)d_in[5];
    const float* W1        = (const float*)d_in[6];
    const float* b1        = (const float*)d_in[7];
    const float* W2        = (const float*)d_in[8];
    const float* b2        = (const float*)d_in[9];
    const float* W3        = (const float*)d_in[10];
    const float* b3        = (const float*)d_in[11];
    float* out = (float*)d_out;

    const int N = in_sizes[0] / NCLS;       // 50000 (multiple of 16)
    const int E = in_sizes[2] / 2;          // 800000
    const int* src = edge;
    const int* dst = edge + E;

    // ---- workspace carve-out (regions overlapped by liveness) ----
    char* p = (char*)d_ws;
    int*    degA = (int*)p;                 p += alignup((size_t)N * 4);
    int*    degD = (int*)p;                 p += alignup((size_t)N * 4);
    float*  dinv = (float*)p;               p += alignup((size_t)N * 4);
    bf16_t* WfT  = (bf16_t*)p;              p += alignup((size_t)FDIM * FHID * 2);
    bf16_t* W1T  = (bf16_t*)p;              p += alignup((size_t)INCH_P * HID * 2);
    bf16_t* W2T  = (bf16_t*)p;              p += alignup((size_t)HID * HID * 2);
    bf16_t* W3T  = (bf16_t*)p;              p += alignup((size_t)HID * OUTP * 2);
    // region A: X0 (bf16 N x 160) ∪ Hlin3 (f32 N x 48)  [X0 dead after conv1 GEMM]
    char* regA = p;                         p += alignup((size_t)N * INCH_P * 2);
    bf16_t* X0    = (bf16_t*)regA;
    float*  Hlin3 = (float*)regA;
    // region B: featBf (bf16 N x 256) ∪ Agg (f32 N x 128) [featBf dead after feat GEMM]
    char* regB = p;                         p += alignup((size_t)N * FDIM * 2);
    bf16_t* featBf = (bf16_t*)regB;
    float*  Agg    = (float*)regB;
    float*  Hlin = (float*)p;               p += alignup((size_t)N * HID * 4);
    bf16_t* Hbf  = (bf16_t*)p;              p += alignup((size_t)N * HID * 2);
    (void)ws_size; (void)n_in; (void)out_size;

    const int B256 = 256;
    const int rt = N / 16;                  // 3125 row tiles

    // 1) degrees
    hipMemsetAsync(degA, 0, (size_t)2 * N * 4, stream);   // degA + degD adjacent
    k_degree<<<(E + B256 - 1) / B256, B256, 0, stream>>>(src, dst, degA, degD, E);
    k_dinv<<<(N + B256 - 1) / B256, B256, 0, stream>>>(degD, dinv, N);

    // 2) converts
    long ftot = (long)N * FDIM;
    k_f2bf<<<(unsigned)((ftot + B256 - 1) / B256), B256, 0, stream>>>(features, featBf, ftot);
    k_wconv<<<(FDIM * FHID + B256 - 1) / B256, B256, 0, stream>>>(Wf, WfT, FDIM, FHID, FDIM, FHID);
    k_wconv<<<(INCH_P * HID + B256 - 1) / B256, B256, 0, stream>>>(W1, W1T, INCH, HID, INCH_P, HID);
    k_wconv<<<(HID * HID + B256 - 1) / B256, B256, 0, stream>>>(W2, W2T, HID, HID, HID, HID);
    k_wconv<<<(HID * OUTP + B256 - 1) / B256, B256, 0, stream>>>(W3, W3T, HID, NCLS, HID, OUTP);

    // 3) assemble X0 (logits / deg-emb / zero pad) + feature projection GEMM into cols 40..103
    long xtot = (long)N * INCH_P;
    k_assemble<<<(unsigned)((xtot + B256 - 1) / B256), B256, 0, stream>>>(logits, deg_table, degA, X0, xtot);
    k_gemm_wide<FDIM, FHID / 16><<<rt, 32, 0, stream>>>(featBf, FDIM, WfT, bf,
                                                        (float*)nullptr, X0, INCH_P, NCLS);

    long htot = (long)N * HID;
    // 4) conv1
    k_gemm_wide<INCH_P, HID / 16><<<rt, 32, 0, stream>>>(X0, INCH_P, W1T, (const float*)nullptr,
                                                         Hlin, (bf16_t*)nullptr, HID, 0);
    hipMemsetAsync(Agg, 0, (size_t)N * HID * 4, stream);
    k_scatter128<<<(unsigned)(((long)E * 32 + B256 - 1) / B256), B256, 0, stream>>>(src, dst, dinv, Hlin, Agg, E);
    k_finalize_hidden<<<(unsigned)((htot + B256 - 1) / B256), B256, 0, stream>>>(Agg, Hlin, dinv, b1, Hbf, htot);

    // 5) conv2
    k_gemm_wide<HID, HID / 16><<<rt, 32, 0, stream>>>(Hbf, HID, W2T, (const float*)nullptr,
                                                      Hlin, (bf16_t*)nullptr, HID, 0);
    hipMemsetAsync(Agg, 0, (size_t)N * HID * 4, stream);
    k_scatter128<<<(unsigned)(((long)E * 32 + B256 - 1) / B256), B256, 0, stream>>>(src, dst, dinv, Hlin, Agg, E);
    k_finalize_hidden<<<(unsigned)((htot + B256 - 1) / B256), B256, 0, stream>>>(Agg, Hlin, dinv, b2, Hbf, htot);

    // 6) conv3 -> d_out
    k_gemm_wide<HID, OUTP / 16><<<rt, 32, 0, stream>>>(Hbf, HID, W3T, (const float*)nullptr,
                                                       Hlin3, (bf16_t*)nullptr, OUTP, 0);
    hipMemsetAsync(out, 0, (size_t)N * NCLS * 4, stream);
    k_scatter40<<<(unsigned)(((long)E * 10 + B256 - 1) / B256), B256, 0, stream>>>(src, dst, dinv, Hlin3, out, E);
    long otot = (long)N * NCLS;
    k_finalize_out<<<(unsigned)((otot + B256 - 1) / B256), B256, 0, stream>>>(Hlin3, dinv, b3, out, otot);
}
